// MultiHeadAttention_11957188952072
// MI455X (gfx1250) — matrix operand order repaired and checked by reading, not compiled, not run on hardware
//
#include <hip/hip_runtime.h>
#include <hip/hip_bf16.h>

typedef __bf16 bf16;
typedef bf16  v16bf __attribute__((ext_vector_type(16)));
typedef bf16  v8bf  __attribute__((ext_vector_type(8)));
typedef float v8f   __attribute__((ext_vector_type(8)));

#define WMMA_BF16(A, Bm, C) \
  __builtin_amdgcn_wmma_f32_16x16x32_bf16(false, (A), false, (Bm), (short)0, (C), false, false)

static __device__ __forceinline__ bf16 f2bf(float f) { return (bf16)f; }

static __device__ __forceinline__ v16bf mk16(v8bf lo, v8bf hi) {
  v16bf r;
#pragma unroll
  for (int i = 0; i < 8; ++i) { r[i] = lo[i]; r[8 + i] = hi[i]; }
  return r;
}

static __device__ __forceinline__ v8bf ld8g(const bf16* p) { return *(const v8bf*)p; }
static __device__ __forceinline__ v8bf ld8s(const bf16* p) { return *(const v8bf*)p; }

static __device__ __forceinline__ v8bf cvt8(const float* p) {
  v8f a = *(const v8f*)p;
  return __builtin_convertvector(a, v8bf);
}

// ---- gfx1250 async global->LDS DMA (ASYNCcnt-tracked), inline asm ----
static __device__ __forceinline__ void async_ld_b128(void* lds, const void* gptr) {
  unsigned lds_off = (unsigned)(size_t)lds;
  asm volatile("global_load_async_to_lds_b128 %0, %1, off"
               :: "v"(lds_off), "v"(gptr) : "memory");
}
static __device__ __forceinline__ void wait_async0() {
  asm volatile("s_wait_asynccnt 0x0" ::: "memory");
}

// ---------------------------------------------------------------------------
// Dims
// ---------------------------------------------------------------------------
constexpr int Bv = 4, Sv = 2048, Dv = 1024, Hv = 16, DEP = 64;
constexpr int Mrows = Bv * Sv;  // 8192

// ---------------------------------------------------------------------------
// Kernel 1: projection  out = (X @ W + bias) * scale
// VTRANS=0: bf16 head-split [B,H,S,64] (Q,K).  VTRANS=1: [B,H,64,S] (V^T).
// Block: 256 threads (8 waves). 128 rows/block; inner loop over 8 N-tiles.
// ---------------------------------------------------------------------------
template <int VTRANS>
__global__ void __launch_bounds__(256)
qkv_proj_kernel(const float* __restrict__ X, const float* __restrict__ W,
                const float* __restrict__ bias, bf16* __restrict__ outHS,
                float scale) {
  __shared__ bf16 WT[64 * 32];  // transposed weight tile: [ncol][k]

  const int tid  = threadIdx.x;
  const int w    = tid >> 5;
  const int lane = tid & 31;
  const int half = lane >> 4;
  const int l16  = lane & 15;

  const int mbase = blockIdx.x * 128;
  const int row   = mbase + w * 16 + l16;
  const float* arow = X + (size_t)row * Dv;

  for (int nb = 0; nb < 8; ++nb) {
    const int nbase = blockIdx.y * 512 + nb * 64;

    v8f acc[4];
#pragma unroll
    for (int t = 0; t < 4; ++t) acc[t] = (v8f)(0.0f);

    for (int k0 = 0; k0 < Dv; k0 += 32) {
      __syncthreads();
      {
        const int off  = tid * 8;
        const int k    = off >> 6;    // 0..31
        const int nloc = off & 63;    // 0..56 step 8
        v8bf wv = cvt8(W + (size_t)(k0 + k) * Dv + nbase + nloc);
#pragma unroll
        for (int i = 0; i < 8; ++i) WT[(nloc + i) * 32 + k] = wv[i];
      }
      __syncthreads();

      v16bf a = mk16(cvt8(arow + k0 + 16 * half), cvt8(arow + k0 + 16 * half + 8));
#pragma unroll
      for (int dt = 0; dt < 4; ++dt) {
        const int ncol = dt * 16 + l16;
        v16bf bm = mk16(ld8s(&WT[ncol * 32 + 16 * half]),
                        ld8s(&WT[ncol * 32 + 16 * half + 8]));
        acc[dt] = WMMA_BF16(a, bm, acc[dt]);
      }
    }

    // epilogue
    const int row0 = mbase + w * 16 + 8 * half;   // first of 8 consecutive rows
    const int bi = row0 >> 11, s0 = row0 & 2047;  // rows don't cross batch
#pragma unroll
    for (int dt = 0; dt < 4; ++dt) {
      const int colg = nbase + dt * 16 + l16;
      const float bb = bias[colg];
      const int h = colg >> 6, d = colg & 63;
      if (VTRANS) {
        // 8 consecutive s positions -> one 16B store
        v8bf pk;
#pragma unroll
        for (int j = 0; j < 8; ++j) pk[j] = f2bf((acc[dt][j] + bb) * scale);
        *(v8bf*)&outHS[(((size_t)bi * Hv + h) * DEP + d) * Sv + s0] = pk;
      } else {
#pragma unroll
        for (int j = 0; j < 8; ++j) {
          outHS[(((size_t)bi * Hv + h) * Sv + s0 + j) * DEP + d] =
              f2bf((acc[dt][j] + bb) * scale);
        }
      }
    }
  }
}

// ---------------------------------------------------------------------------
// Kernel 2: flash attention, 64-key tiles, async double-buffered staging.
// Qh/Kh: bf16 [B,H,S,64].  Vg: bf16 [B,H,64,S] (pre-transposed).
// Grid: (S/128, H, B). Block 256 (8 waves), wave = 16 query rows.
// ---------------------------------------------------------------------------
__global__ void __launch_bounds__(256)
flash_attn_kernel(const bf16* __restrict__ Qh, const bf16* __restrict__ Kh,
                  const bf16* __restrict__ Vg, const float* __restrict__ mask,
                  bf16* __restrict__ ctx) {
  __shared__ bf16 Kt[2][64 * 64];     // [buf][key][d]       (2 x 8 KB)
  __shared__ bf16 Vt[2][64 * 64];     // [buf][d][key]       (2 x 8 KB)
  __shared__ bf16 Pls[8 * 16 * 64];   // per-wave P tile     (16 KB)

  const int tid  = threadIdx.x;
  const int w    = tid >> 5;
  const int lane = tid & 31;
  const int half = lane >> 4;
  const int l16  = lane & 15;

  const int b = blockIdx.z, h = blockIdx.y;
  const int qbase = blockIdx.x * 128 + w * 16;
  const size_t headoff  = ((size_t)b * Hv + h) * Sv;
  const size_t vheadoff = ((size_t)b * Hv + h) * DEP;

  const int srow = tid >> 2;          // 0..63
  const int sseg = (tid & 3) * 16;    // 0,16,32,48 elements

  const bf16* qrow = Qh + (headoff + qbase + l16) * DEP;
  const v16bf aq0 = mk16(ld8g(qrow + 16 * half),      ld8g(qrow + 16 * half + 8));
  const v16bf aq1 = mk16(ld8g(qrow + 32 + 16 * half), ld8g(qrow + 32 + 16 * half + 8));

  v8f acc[4];
#pragma unroll
  for (int t = 0; t < 4; ++t) acc[t] = (v8f)(0.0f);
  float mrow[8], lrow[8];
#pragma unroll
  for (int j = 0; j < 8; ++j) { mrow[j] = -1e30f; lrow[j] = 0.0f; }

  constexpr int NT = Sv / 64;  // 32 key tiles

  auto stage = [&](int kt, int buf) {
    const int kbase = kt * 64;
    const bf16* gK = Kh + (headoff + kbase + srow) * DEP + sseg;
    async_ld_b128(&Kt[buf][srow * 64 + sseg],     gK);
    async_ld_b128(&Kt[buf][srow * 64 + sseg + 8], gK + 8);
    const bf16* gV = Vg + (vheadoff + srow) * Sv + kbase + sseg;
    async_ld_b128(&Vt[buf][srow * 64 + sseg],     gV);
    async_ld_b128(&Vt[buf][srow * 64 + sseg + 8], gV + 8);
  };

  stage(0, 0);

  for (int kt = 0; kt < NT; ++kt) {
    const int kbase = kt * 64;
    const int buf = kt & 1;

    wait_async0();      // my DMAs for tile kt complete
    __syncthreads();    // everyone's DMAs done; other buffer free

    if (kt + 1 < NT) stage(kt + 1, buf ^ 1);

    // ---- logits: four 16-key subtiles, 2 WMMAs each over depth ----
    v8f c[4];
#pragma unroll
    for (int sub = 0; sub < 4; ++sub) {
      c[sub] = (v8f)(0.0f);
      const int key = sub * 16 + l16;
      v16bf b0 = mk16(ld8s(&Kt[buf][key * 64 + 16 * half]),
                      ld8s(&Kt[buf][key * 64 + 16 * half + 8]));
      v16bf b1 = mk16(ld8s(&Kt[buf][key * 64 + 32 + 16 * half]),
                      ld8s(&Kt[buf][key * 64 + 32 + 16 * half + 8]));
      c[sub] = WMMA_BF16(aq0, b0, c[sub]);
      c[sub] = WMMA_BF16(aq1, b1, c[sub]);
      const float ml = mask[(size_t)b * Sv + kbase + key] * -1e9f;
#pragma unroll
      for (int j = 0; j < 8; ++j) c[sub][j] += ml;
    }

    // ---- online softmax across 64 keys ----
#pragma unroll
    for (int j = 0; j < 8; ++j) {
      float mx = fmaxf(fmaxf(c[0][j], c[1][j]), fmaxf(c[2][j], c[3][j]));
      mx = fmaxf(mx, __shfl_xor(mx, 1, 32));
      mx = fmaxf(mx, __shfl_xor(mx, 2, 32));
      mx = fmaxf(mx, __shfl_xor(mx, 4, 32));
      mx = fmaxf(mx, __shfl_xor(mx, 8, 32));
      const float newm  = fmaxf(mrow[j], mx);
      const float alpha = __expf(mrow[j] - newm);
      float e0 = __expf(c[0][j] - newm);
      float e1 = __expf(c[1][j] - newm);
      float e2 = __expf(c[2][j] - newm);
      float e3 = __expf(c[3][j] - newm);
      float s = (e0 + e1) + (e2 + e3);
      s += __shfl_xor(s, 1, 32);
      s += __shfl_xor(s, 2, 32);
      s += __shfl_xor(s, 4, 32);
      s += __shfl_xor(s, 8, 32);
      lrow[j] = lrow[j] * alpha + s;
      mrow[j] = newm;
#pragma unroll
      for (int t = 0; t < 4; ++t) acc[t][j] *= alpha;
      bf16* prow = &Pls[(w * 16 + j + 8 * half) * 64];
      prow[l16]      = f2bf(e0);
      prow[16 + l16] = f2bf(e1);
      prow[32 + l16] = f2bf(e2);
      prow[48 + l16] = f2bf(e3);
    }

    // ---- P @ V: K-dim = 64 keys -> 2 WMMAs per 16-col n-tile ----
    const bf16* pr = &Pls[(w * 16 + l16) * 64];
    v16bf ap0 = mk16(ld8s(pr + 16 * half),      ld8s(pr + 16 * half + 8));
    v16bf ap1 = mk16(ld8s(pr + 32 + 16 * half), ld8s(pr + 32 + 16 * half + 8));
#pragma unroll
    for (int dt = 0; dt < 4; ++dt) {
      const int col = dt * 16 + l16;
      v16bf bv0 = mk16(ld8s(&Vt[buf][col * 64 + 16 * half]),
                       ld8s(&Vt[buf][col * 64 + 16 * half + 8]));
      v16bf bv1 = mk16(ld8s(&Vt[buf][col * 64 + 32 + 16 * half]),
                       ld8s(&Vt[buf][col * 64 + 32 + 16 * half + 8]));
      acc[dt] = WMMA_BF16(ap0, bv0, acc[dt]);
      acc[dt] = WMMA_BF16(ap1, bv1, acc[dt]);
    }
  }

  // epilogue: normalize, store bf16 context [B,S,1024]
#pragma unroll
  for (int j = 0; j < 8; ++j) {
    const float inv = 1.0f / lrow[j];
    const int rowg = qbase + j + 8 * half;
#pragma unroll
    for (int dt = 0; dt < 4; ++dt) {
      const int colg = h * DEP + dt * 16 + l16;
      ctx[((size_t)b * Sv + rowg) * Dv + colg] = f2bf(acc[dt][j] * inv);
    }
  }
}

// ---------------------------------------------------------------------------
// Kernel 3: output projection  out = ctx @ wo + bo   (f32 output)
// ---------------------------------------------------------------------------
__global__ void __launch_bounds__(256)
out_proj_kernel(const bf16* __restrict__ A, const float* __restrict__ W,
                const float* __restrict__ bias, float* __restrict__ out) {
  __shared__ bf16 WT[64 * 32];

  const int tid  = threadIdx.x;
  const int w    = tid >> 5;
  const int lane = tid & 31;
  const int half = lane >> 4;
  const int l16  = lane & 15;

  const int mbase = blockIdx.x * 128;
  const int row   = mbase + w * 16 + l16;
  const bf16* arow = A + (size_t)row * Dv;

  for (int nb = 0; nb < 8; ++nb) {
    const int nbase = blockIdx.y * 512 + nb * 64;

    v8f acc[4];
#pragma unroll
    for (int t = 0; t < 4; ++t) acc[t] = (v8f)(0.0f);

    for (int k0 = 0; k0 < Dv; k0 += 32) {
      __syncthreads();
      {
        const int off  = tid * 8;
        const int k    = off >> 6;
        const int nloc = off & 63;
        v8bf wv = cvt8(W + (size_t)(k0 + k) * Dv + nbase + nloc);
#pragma unroll
        for (int i = 0; i < 8; ++i) WT[(nloc + i) * 32 + k] = wv[i];
      }
      __syncthreads();

      v16bf a = mk16(ld8g(arow + k0 + 16 * half), ld8g(arow + k0 + 16 * half + 8));
#pragma unroll
      for (int dt = 0; dt < 4; ++dt) {
        const int ncol = dt * 16 + l16;
        v16bf bm = mk16(ld8s(&WT[ncol * 32 + 16 * half]),
                        ld8s(&WT[ncol * 32 + 16 * half + 8]));
        acc[dt] = WMMA_BF16(a, bm, acc[dt]);
      }
    }

#pragma unroll
    for (int dt = 0; dt < 4; ++dt) {
      const int colg = nbase + dt * 16 + l16;
      const float bb = bias[colg];
#pragma unroll
      for (int j = 0; j < 8; ++j) {
        const int rowg = mbase + w * 16 + j + 8 * half;
        out[(size_t)rowg * Dv + colg] = acc[dt][j] + bb;
      }
    }
  }
}

// ---------------------------------------------------------------------------
// Host launcher
// ---------------------------------------------------------------------------
extern "C" void kernel_launch(void* const* d_in, const int* in_sizes, int n_in,
                              void* d_out, int out_size, void* d_ws, size_t ws_size,
                              hipStream_t stream) {
  (void)in_sizes; (void)n_in; (void)out_size; (void)ws_size;

  const float* q    = (const float*)d_in[0];
  const float* k    = (const float*)d_in[1];
  const float* v    = (const float*)d_in[2];
  const float* mask = (const float*)d_in[3];
  const float* wq   = (const float*)d_in[4];
  const float* bq   = (const float*)d_in[5];
  const float* wk   = (const float*)d_in[6];
  const float* bk   = (const float*)d_in[7];
  const float* wv   = (const float*)d_in[8];
  const float* bv   = (const float*)d_in[9];
  const float* wo   = (const float*)d_in[10];
  const float* bo   = (const float*)d_in[11];
  float* out = (float*)d_out;

  const size_t seg = (size_t)Bv * Sv * Dv * sizeof(bf16);  // 16 MiB
  char* ws = (char*)d_ws;
  bf16* Qh  = (bf16*)(ws);
  bf16* Kh  = (bf16*)(ws + seg);
  bf16* Vg  = (bf16*)(ws + 2 * seg);   // [B,H,64,S] (V^T)
  bf16* ctx = (bf16*)(ws + 3 * seg);

  const dim3 gblk(Mrows / 128, 2);
  const dim3 tblk(256);
  const float qscale = 0.125f;  // 1/sqrt(64)

  qkv_proj_kernel<0><<<gblk, tblk, 0, stream>>>(q, wq, bq, Qh, qscale);
  qkv_proj_kernel<0><<<gblk, tblk, 0, stream>>>(k, wk, bk, Kh, 1.0f);
  qkv_proj_kernel<1><<<gblk, tblk, 0, stream>>>(v, wv, bv, Vg, 1.0f);

  flash_attn_kernel<<<dim3(Sv / 128, Hv, Bv), tblk, 0, stream>>>(Qh, Kh, Vg, mask, ctx);

  out_proj_kernel<<<gblk, tblk, 0, stream>>>(ctx, wo, bo, out);
}
